// TransformerBlock_8727373545599
// MI455X (gfx1250) — matrix operand order repaired and checked by reading, not compiled, not run on hardware
//
#include <hip/hip_runtime.h>
#include <hip/hip_bf16.h>
#include <cstdint>

typedef __attribute__((ext_vector_type(16))) _Float16 v16h;
typedef __attribute__((ext_vector_type(8)))  float    v8f;
typedef __attribute__((ext_vector_type(4)))  unsigned int u32x4;
typedef __attribute__((ext_vector_type(4)))  int      i32x4;
typedef __attribute__((ext_vector_type(8)))  int      i32x8;

#if defined(__has_builtin)
#  if __has_builtin(__builtin_amdgcn_tensor_load_to_lds)
#    define HAVE_TDM 1
#  endif
#endif
#ifndef HAVE_TDM
#  define HAVE_TDM 0
#endif

// ---------------- WMMA helpers (CDNA5 gfx1250, wave32) ----------------
__device__ __forceinline__ v8f wmma32(v16h a, v16h b, v8f c) {
  return __builtin_amdgcn_wmma_f32_16x16x32_f16(false, a, false, b, (short)0, c,
                                                false, false);
}

// A fragment: 16x32 f16. lane&15 = M, lane>>4 selects K-half per ISA table.
__device__ __forceinline__ v16h load_a16(const _Float16* base, int row0, int k0,
                                         int ld, int lane) {
  v16h a;
  int m = lane & 15;
  int kb = (lane >> 4) << 3;
  const _Float16* r = base + (size_t)(row0 + m) * ld + k0;
#pragma unroll
  for (int i = 0; i < 16; ++i) {
    int p = i >> 1;
    int k = ((p >> 2) << 4) + kb + ((p & 3) << 1) + (i & 1);
    a[i] = r[k];
  }
  return a;
}

// B fragment: 32x16 f16, element (k,n). Storage row-major [k][n], stride ld.
// lane&15 = N, lane>>4 selects K-half per ISA table.
__device__ __forceinline__ v16h load_b_rm(const _Float16* base, int k0, int n0,
                                          int ld, int lane) {
  v16h b;
  int kb = (lane >> 4) << 3;
  const _Float16* r = base + (size_t)k0 * ld + n0 + (lane & 15);
#pragma unroll
  for (int i = 0; i < 16; ++i) b[i] = r[(size_t)(((i >> 3) << 4) + kb + (i & 7)) * ld];
  return b;
}

// ---------------- TDM: 1-D contiguous tile, global -> LDS ----------------
__device__ __forceinline__ void tdm_load_1d_to_lds(const void* gptr,
                                                   unsigned lds_off,
                                                   unsigned nbytes) {
#if HAVE_TDM
  unsigned long long ga = (unsigned long long)(uintptr_t)gptr;
  unsigned n4 = nbytes >> 2;  // dword elements
  u32x4 g0;
  g0[0] = 1u;                                   // count=1 valid descriptor
  g0[1] = lds_off;                              // lds_addr
  g0[2] = (unsigned)(ga & 0xFFFFFFFFu);         // global_addr[31:0]
  g0[3] = (unsigned)((ga >> 32) & 0x01FFFFFFu) | (2u << 30);  // [56:32]|type=2
  i32x8 g1;
  g1[0] = (int)(2u << 16);                      // data_size=2 (4B), mask=0
  g1[1] = (int)((n4 & 0xFFFFu) << 16);          // tensor_dim0[15:0] @ 63:48
  g1[2] = (int)((n4 >> 16) & 0xFFFFu);          // tensor_dim0[31:16] @ 79:64
  g1[3] = (int)((n4 & 0xFFFFu) << 16);          // tile_dim0 @ 127:112
  g1[4] = 0;                                    // tile_dim1/2 = 0
  g1[5] = (int)n4;                              // tensor_dim0_stride[31:0]
  g1[6] = 0;
  g1[7] = 0;
  i32x4 z4 = {0, 0, 0, 0};
#if __clang_major__ >= 23
  i32x8 z8 = {0, 0, 0, 0, 0, 0, 0, 0};
  __builtin_amdgcn_tensor_load_to_lds(g0, g1, z4, z4, z8, 0);
#else
  __builtin_amdgcn_tensor_load_to_lds(g0, g1, z4, z4, 0);
#endif
#endif
}

__device__ __forceinline__ void tdm_wait0() {
#if defined(__has_builtin)
#if __has_builtin(__builtin_amdgcn_s_wait_tensorcnt)
  __builtin_amdgcn_s_wait_tensorcnt(0);
#endif
#endif
}

// ---------------- constants ----------------
#define CC   180
#define HPAD 192     // 6 heads * 32 (head dim 30 padded to 32)
#define NW   256
#define NTOK 256
#define NG   64      // permuted kv tokens per window
#define QSCALE 0.18257418583505536f  // 30^-0.5

// workspace offsets (bytes)
#define OFF_XH     ((size_t)0)            // f16 1024*256*192  (reused as yf)
#define OFF_BIAS   ((size_t)100663296)    // f32 6*256*64
#define OFF_WQT    ((size_t)101056512)    // f16 192*192
#define OFF_BQ     ((size_t)101130240)    // f32 192
#define OFF_WKVT   ((size_t)101131008)    // f16 192*96
#define OFF_BKV    ((size_t)101167872)    // f32 96
#define OFF_WPROJT ((size_t)101168256)    // f16 192*192
#define OFF_WPIN   ((size_t)101241984)    // f16 192*720
#define OFF_WPOUT  ((size_t)101518464)    // f16 384*192
#define OFF_FILMA  ((size_t)101665920)    // f32 4*360
#define OFF_FILMF  ((size_t)101671680)    // f32 4*360

// ---------------- kernel 0: weight prep / film / bias table ----------------
__global__ void prep_kernel(const float* k_v, const float* w_kern_attn,
                            const float* w_kern_ffn, const float* rel_table,
                            const float* w_q, const float* b_q,
                            const float* w_kv, const float* b_kv,
                            const float* w_proj, const float* w_pin,
                            const float* w_pout,
                            _Float16* WqT, float* bq_pad, _Float16* WkvT,
                            float* bkv_pad, _Float16* WprojT, _Float16* WpinT,
                            _Float16* WpoutT, float* filmA, float* filmF,
                            float* biasT) {
  int gid = blockIdx.x * blockDim.x + threadIdx.x;
  int nthr = gridDim.x * blockDim.x;
  // FiLM vectors: (2, B=4, 360) dots of length 256
  for (int j = gid; j < 2880; j += nthr) {
    int which = j / 1440, r = j % 1440, b = r / 360, o = r % 360;
    const float* w = which ? w_kern_ffn : w_kern_attn;
    float s = 0.f;
    for (int i = 0; i < 256; ++i) s += k_v[b * 256 + i] * w[o * 256 + i];
    (which ? filmF : filmA)[b * 360 + o] = s;
  }
  // WqT: (192 x 192), out col n -> head n/32, dim n%32; pre-scaled
  for (int j = gid; j < 192 * 192; j += nthr) {
    int k = j / 192, n = j % 192, h = n >> 5, d = n & 31;
    float v = (d < 30 && k < 180) ? w_q[(h * 30 + d) * 180 + k] * QSCALE : 0.f;
    WqT[j] = (_Float16)v;
  }
  for (int n = gid; n < 192; n += nthr) {
    int h = n >> 5, d = n & 31;
    bq_pad[n] = (d < 30) ? b_q[h * 30 + d] * QSCALE : 0.f;
  }
  // WkvT: (192 x 96)
  for (int j = gid; j < 192 * 96; j += nthr) {
    int k = j / 96, n = j % 96;
    WkvT[j] = (_Float16)((n < 90 && k < 180) ? w_kv[n * 180 + k] : 0.f);
  }
  for (int n = gid; n < 96; n += nthr) bkv_pad[n] = (n < 90) ? b_kv[n] : 0.f;
  // WprojT: (192 x 192), in row k -> orig channel h*30+d
  for (int j = gid; j < 192 * 192; j += nthr) {
    int k = j / 192, n = j % 192, h = k >> 5, d = k & 31;
    float v = (d < 30 && n < 180) ? w_proj[n * 180 + (h * 30 + d)] : 0.f;
    WprojT[j] = (_Float16)v;
  }
  // WpinT: (192 x 720)
  for (int j = gid; j < 192 * 720; j += nthr) {
    int k = j / 720, n = j % 720;
    WpinT[j] = (_Float16)((k < 180) ? w_pin[n * 180 + k] : 0.f);
  }
  // WpoutT: (384 x 192)
  for (int j = gid; j < 384 * 192; j += nthr) {
    int k = j / 192, n = j % 192;
    WpoutT[j] = (_Float16)((k < 360 && n < 180) ? w_pout[n * 360 + k] : 0.f);
  }
  // bias table: bias[h][t][g] = rel_table[idx][h]
  for (int j = gid; j < 6 * 256 * 64; j += nthr) {
    int h = j / (256 * 64), t = (j >> 6) & 255, g = j & 63;
    int rh = t >> 5, rw = (t & 15) >> 1;  // (row/2, col/2)
    int gh = g >> 3, gw = g & 7;
    int idx = (rh - gh + 7) * 15 + (rw - gw + 7);
    biasT[j] = rel_table[idx * 6 + h];
  }
}

// ---------------- kernel A/C: LayerNorm + FiLM ----------------
// mode 0: write windowed layout xh[win][tok][192]; mode 1: pixel-major yf
template <int MODE>
__global__ __launch_bounds__(256) void ln_film_kernel(
    const float* x, const float* lnw, const float* lnb, const float* film,
    _Float16* dst0) {
  int p = blockIdx.x * 256 + threadIdx.x;  // 0..262143 pixel id
  int b = p >> 16, rem = p & 65535, hh = rem >> 8, ww = rem & 255;
  const float* xb = x + (size_t)b * CC * 65536 + hh * 256 + ww;
  float s = 0.f, s2 = 0.f;
  for (int c = 0; c < CC; ++c) {
    float v = xb[(size_t)c * 65536];
    s += v; s2 += v * v;
  }
  float mu = s * (1.f / CC);
  float var = s2 * (1.f / CC) - mu * mu;
  float rstd = rsqrtf(var + 1e-5f);
  _Float16* dst;
  if (MODE == 0) {
    int win = b * NW + (hh >> 4) * 16 + (ww >> 4);
    int tok = (hh & 15) * 16 + (ww & 15);
    dst = dst0 + (size_t)win * (NTOK * HPAD) + tok * HPAD;
  } else {
    dst = dst0 + (size_t)p * HPAD;
  }
  const float* f = film + b * 360;
  for (int c = 0; c < CC; ++c) {
    float v = xb[(size_t)c * 65536];
    float yn = (v - mu) * rstd * lnw[c] + lnb[c];
    dst[c] = (_Float16)(yn * f[c] + f[180 + c]);
  }
  for (int c = CC; c < HPAD; ++c) dst[c] = (_Float16)0.f;
}

// ---------------- kernel B: per-window permuted-window attention ----------------
// LDS map (288KB dynamic):
//   [0      , 98304 )  Xs 256x192 f16   (reused as Os)
//   [98304  , 196608)  Qs 256x192 f16
//   [196608 , 245760)  Kvp 256x96 f16   (reused: Ps 256x64 f16 @196608,
//                                        sums 256 f32 @229376,
//                                        store-scratch 8x256 f32 @230400)
//   [245760 , 270336)  KsT 192x64 f16   (K transposed: [head*32+d][g])
//   [270336 , 294912)  Vs  64x192 f16
__global__ __launch_bounds__(256) void attn_kernel(
    const _Float16* xh, const _Float16* WqT, const float* bq_pad,
    const _Float16* WkvT, const float* bkv_pad, const _Float16* WprojT,
    const float* b_proj, const float* biasT, const float* x, float* out) {
  extern __shared__ char smem[];
  _Float16* Xs   = (_Float16*)smem;
  _Float16* Qs   = (_Float16*)(smem + 98304);
  _Float16* Kvp  = (_Float16*)(smem + 196608);
  float*    sums = (float*)(smem + 229376);
  float*    scr  = (float*)(smem + 230400);
  _Float16* KsT  = (_Float16*)(smem + 245760);
  _Float16* Vs   = (_Float16*)(smem + 270336);
  _Float16* Ps   = Kvp;
  _Float16* Os   = Xs;

  const int tid = threadIdx.x, lane = tid & 31, wave = tid >> 5;
  const int win = blockIdx.x;

  // B1: stage window tile into LDS via Tensor Data Mover (96KB contiguous)
#if HAVE_TDM
  if (wave == 0) {
    tdm_load_1d_to_lds(xh + (size_t)win * (NTOK * HPAD), 0u,
                       NTOK * HPAD * 2u);
    tdm_wait0();
  }
#else
  {
    const uint4* src = (const uint4*)(xh + (size_t)win * (NTOK * HPAD));
    uint4* d = (uint4*)Xs;
    for (int i = tid; i < 6144; i += 256) d[i] = src[i];
  }
#endif
  __builtin_prefetch(WqT, 0, 1);
  __builtin_prefetch(WkvT, 0, 1);
  __syncthreads();

  // B2: Q = X*WqT + bq (pre-scaled), KVp = X*WkvT + bkv.  192+96 tiles / 8 waves
  for (int ti = wave * 36; ti < wave * 36 + 36; ++ti) {
    if (ti < 192) {
      int tm = ti / 12, tn = ti % 12;
      float bb = bq_pad[tn * 16 + (lane & 15)];
      v8f acc = {bb, bb, bb, bb, bb, bb, bb, bb};
#pragma unroll
      for (int ks = 0; ks < 6; ++ks) {
        v16h a = load_a16(Xs, tm * 16, ks * 32, HPAD, lane);
        v16h bf = load_b_rm(WqT, ks * 32, tn * 16, 192, lane);
        acc = wmma32(a, bf, acc);
      }
      int mhi = (lane >> 4) << 3, n = tn * 16 + (lane & 15);
#pragma unroll
      for (int r = 0; r < 8; ++r)
        Qs[(tm * 16 + r + mhi) * HPAD + n] = (_Float16)acc[r];
    } else {
      int t2 = ti - 192, tm = t2 / 6, tn = t2 % 6;
      float bb = bkv_pad[tn * 16 + (lane & 15)];
      v8f acc = {bb, bb, bb, bb, bb, bb, bb, bb};
#pragma unroll
      for (int ks = 0; ks < 6; ++ks) {
        v16h a = load_a16(Xs, tm * 16, ks * 32, HPAD, lane);
        v16h bf = load_b_rm(WkvT, ks * 32, tn * 16, 96, lane);
        acc = wmma32(a, bf, acc);
      }
      int mhi = (lane >> 4) << 3, n = tn * 16 + (lane & 15);
#pragma unroll
      for (int r = 0; r < 8; ++r)
        Kvp[(tm * 16 + r + mhi) * 96 + n] = (_Float16)acc[r];
    }
  }
  __syncthreads();

  // B3: 2x2 pixel-group -> channel permutation gather.
  // K is written transposed (KsT[d_padded][g]) for contiguous B-fragment loads.
  for (int e = tid; e < 2 * NG * HPAD; e += 256) {
    int which = e / (NG * HPAD);
    int r = (e - which * NG * HPAD) / HPAD;   // kv-group id 0..63
    int n = e % HPAD;                          // padded feature h*32+d
    int h = n >> 5, d = n & 31;
    _Float16 v = (_Float16)0.f;
    if (d < 30) {
      int f = h * 30 + d;
      int rp = f / 90, cp = (f / 45) & 1, ch = (f % 45) + which * 45;
      int t = ((r >> 3) * 2 + rp) * 16 + ((r & 7) * 2 + cp);
      v = Kvp[t * 96 + ch];
    }
    if (which) Vs[r * HPAD + n] = v;
    else       KsT[n * NG + r] = v;
  }
  __syncthreads();

  // B4: per-head attention (softmax over 64 kv-groups)
  for (int h = 0; h < 6; ++h) {
    v8f S[2][4];
#pragma unroll
    for (int tmi = 0; tmi < 2; ++tmi) {
      int tm = wave * 2 + tmi;
      v16h a = load_a16(Qs, tm * 16, h * 32, HPAD, lane);
#pragma unroll
      for (int tn = 0; tn < 4; ++tn) {
        v16h bf = load_b_rm(KsT, h * 32, tn * 16, NG, lane);
        v8f z = {0.f, 0.f, 0.f, 0.f, 0.f, 0.f, 0.f, 0.f};
        v8f sv = wmma32(a, bf, z);
        int g = tn * 16 + (lane & 15);
        int mhi = (lane >> 4) << 3;
#pragma unroll
        for (int r = 0; r < 8; ++r)
          sv[r] += biasT[(h * 256 + tm * 16 + r + mhi) * 64 + g];
        S[tmi][tn] = sv;
      }
    }
    // row softmax: each 16-lane half holds one row across the 4 N-tiles
#pragma unroll
    for (int tmi = 0; tmi < 2; ++tmi) {
      int tm = wave * 2 + tmi;
      int mhi = (lane >> 4) << 3, gb = lane & 15;
#pragma unroll
      for (int r = 0; r < 8; ++r) {
        float mx = fmaxf(fmaxf(S[tmi][0][r], S[tmi][1][r]),
                         fmaxf(S[tmi][2][r], S[tmi][3][r]));
#pragma unroll
        for (int msk = 1; msk < 16; msk <<= 1)
          mx = fmaxf(mx, __shfl_xor(mx, msk, 32));
        float e0 = __expf(S[tmi][0][r] - mx), e1 = __expf(S[tmi][1][r] - mx);
        float e2 = __expf(S[tmi][2][r] - mx), e3 = __expf(S[tmi][3][r] - mx);
        float sm = e0 + e1 + e2 + e3;
#pragma unroll
        for (int msk = 1; msk < 16; msk <<= 1) sm += __shfl_xor(sm, msk, 32);
        int tok = tm * 16 + r + mhi;
        Ps[tok * 64 +  0 + gb] = (_Float16)e0;
        Ps[tok * 64 + 16 + gb] = (_Float16)e1;
        Ps[tok * 64 + 32 + gb] = (_Float16)e2;
        Ps[tok * 64 + 48 + gb] = (_Float16)e3;
        sums[tok] = sm;
      }
    }
    __syncthreads();
    // O_h = P @ V_h  (256x64 @ 64x32), normalize by row sums
#pragma unroll
    for (int j = 0; j < 4; ++j) {
      int ti = wave * 4 + j, tm = ti >> 1, tn = ti & 1;
      v8f acc = {0.f, 0.f, 0.f, 0.f, 0.f, 0.f, 0.f, 0.f};
#pragma unroll
      for (int ks = 0; ks < 2; ++ks) {
        v16h a = load_a16(Ps, tm * 16, ks * 32, 64, lane);
        v16h bf = load_b_rm(Vs, ks * 32, h * 32 + tn * 16, HPAD, lane);
        acc = wmma32(a, bf, acc);
      }
      int mhi = (lane >> 4) << 3, nn = h * 32 + tn * 16 + (lane & 15);
#pragma unroll
      for (int r = 0; r < 8; ++r) {
        int tok = tm * 16 + r + mhi;
        Os[tok * HPAD + nn] = (_Float16)(acc[r] / sums[tok]);
      }
    }
    __syncthreads();
  }

  // B5: proj + residual; transpose each D-tile through per-wave LDS scratch
  // so global stores are 16 consecutive floats per lane group.
  int bb_ = win >> 8, wi = win & 255, whh = wi >> 4, www = wi & 15;
  float* wscr = scr + wave * 256;
  __builtin_prefetch(WprojT, 0, 1);
  for (int j = 0; j < 24; ++j) {
    int ti = wave * 24 + j, tm = ti / 12, tn = ti % 12;
    int colb = tn * 16 + (lane & 15);
    float bb = (colb < 180) ? b_proj[colb] : 0.f;
    v8f acc = {bb, bb, bb, bb, bb, bb, bb, bb};
#pragma unroll
    for (int ks = 0; ks < 6; ++ks) {
      v16h a = load_a16(Os, tm * 16, ks * 32, HPAD, lane);
      v16h bf = load_b_rm(WprojT, ks * 32, tn * 16, 192, lane);
      acc = wmma32(a, bf, acc);
    }
    int mhi = (lane >> 4) << 3;
#pragma unroll
    for (int r = 0; r < 8; ++r)
      wscr[(lane & 15) * 16 + (r + mhi)] = acc[r];  // [col_local][m_local]
    // within this tile: hh is constant (= whh*16 + tm)
    int hh = whh * 16 + tm;
    for (int idx = lane; idx < 256; idx += 32) {
      int cl = idx >> 4, m = idx & 15;
      int col = tn * 16 + cl;
      if (col < 180) {
        size_t gi = (((size_t)bb_ * CC + col) * 256 + hh) * 256 + www * 16 + m;
        out[gi] = x[gi] + wscr[idx];
      }
    }
  }
}

// ---------------- kernel D: fused FFN (pin -> dw3x3 -> gelu gate -> pout) ----------------
// LDS map (255.5KB dynamic):
//   [0      , 43008 )  Ys 112x192 f16
//   [43008  , 204288)  T  112x720 f16
//   [204288 , 253440)  G  64x384 f16
//   [253440 , 261632)  store-scratch 8x256 f32
__global__ __launch_bounds__(256) void ffn_kernel(
    const _Float16* yf, const _Float16* WpinT, const float* w_dw,
    const _Float16* WpoutT, float* out) {
  extern __shared__ char smem[];
  _Float16* Ys = (_Float16*)smem;
  _Float16* T  = (_Float16*)(smem + 43008);
  _Float16* G  = (_Float16*)(smem + 204288);
  float*    scr = (float*)(smem + 253440);
  int tid = threadIdx.x, lane = tid & 31, wave = tid >> 5;
  int blk = blockIdx.x;
  int b = blk >> 10, t = blk & 1023, ty = t >> 5, tx = t & 31;
  int y0 = ty * 8, x0 = tx * 8;

  // D1: load 10x10 halo (zero-padded borders), rows padded to 112
  for (int e = tid; e < 112 * HPAD; e += 256) {
    int p = e / HPAD, c = e % HPAD;
    _Float16 v = (_Float16)0.f;
    if (p < 100) {
      int iy = p / 10, ix = p % 10;
      int gy = y0 - 1 + iy, gx = x0 - 1 + ix;
      if (gy >= 0 && gy < 256 && gx >= 0 && gx < 256)
        v = yf[((size_t)((b * 256 + gy) * 256 + gx)) * HPAD + c];
    }
    Ys[e] = v;
  }
  __builtin_prefetch(WpinT, 0, 1);
  __syncthreads();

  // D2: T = Ys @ WpinT  (112x192 @ 192x720)
  for (int ti = wave; ti < 7 * 45; ti += 8) {
    int tm = ti / 45, tn = ti % 45;
    v8f acc = {0.f, 0.f, 0.f, 0.f, 0.f, 0.f, 0.f, 0.f};
#pragma unroll
    for (int ks = 0; ks < 6; ++ks) {
      v16h a = load_a16(Ys, tm * 16, ks * 32, HPAD, lane);
      v16h bf = load_b_rm(WpinT, ks * 32, tn * 16, 720, lane);
      acc = wmma32(a, bf, acc);
    }
    int mhi = (lane >> 4) << 3, n = tn * 16 + (lane & 15);
#pragma unroll
    for (int r = 0; r < 8; ++r)
      T[(tm * 16 + r + mhi) * 720 + n] = (_Float16)acc[r];
  }
  __syncthreads();

  // D3: depthwise 3x3 on both halves, exact-GELU gate
  for (int e = tid; e < 64 * 360; e += 256) {
    int q = e / 360, c = e % 360;
    int oy = q >> 3, ox = q & 7;
    float sa = 0.f, sb = 0.f;
#pragma unroll
    for (int dy = 0; dy < 3; ++dy)
#pragma unroll
      for (int dx = 0; dx < 3; ++dx) {
        int p = (oy + dy) * 10 + (ox + dx);
        sa += (float)T[p * 720 + c] * w_dw[c * 9 + dy * 3 + dx];
        sb += (float)T[p * 720 + c + 360] * w_dw[(c + 360) * 9 + dy * 3 + dx];
      }
    float g = 0.5f * sa * (1.f + erff(sa * 0.70710678118654752f));
    G[q * 384 + c] = (_Float16)(g * sb);
  }
  for (int e = tid; e < 64 * 24; e += 256)
    G[(e / 24) * 384 + 360 + (e % 24)] = (_Float16)0.f;
  __syncthreads();

  // D4: out += G @ WpoutT (64x384 @ 384x192), staged transpose for coalescing
  float* wscr = scr + wave * 256;
  for (int j = 0; j < 6; ++j) {
    int ti = wave * 6 + j, tm = ti / 12, tn = ti % 12;
    v8f acc = {0.f, 0.f, 0.f, 0.f, 0.f, 0.f, 0.f, 0.f};
#pragma unroll
    for (int ks = 0; ks < 12; ++ks) {
      v16h a = load_a16(G, tm * 16, ks * 32, 384, lane);
      v16h bf = load_b_rm(WpoutT, ks * 32, tn * 16, 192, lane);
      acc = wmma32(a, bf, acc);
    }
    int mhi = (lane >> 4) << 3;
#pragma unroll
    for (int r = 0; r < 8; ++r)
      wscr[(lane & 15) * 16 + (r + mhi)] = acc[r];  // [col_local][m_local]
    for (int idx = lane; idx < 256; idx += 32) {
      int cl = idx >> 4, m = idx & 15;
      int col = tn * 16 + cl;
      if (col < 180) {
        int q = tm * 16 + m;  // 0..63
        int gy = y0 + (q >> 3), gx = x0 + (q & 7);
        size_t gi = (((size_t)b * CC + col) * 256 + gy) * 256 + gx;
        out[gi] += wscr[idx];
      }
    }
  }
}

// ---------------- launcher ----------------
extern "C" void kernel_launch(void* const* d_in, const int* in_sizes, int n_in,
                              void* d_out, int out_size, void* d_ws,
                              size_t ws_size, hipStream_t stream) {
  (void)in_sizes; (void)n_in; (void)out_size; (void)ws_size;
  const float* x           = (const float*)d_in[0];
  const float* k_v         = (const float*)d_in[1];
  const float* ln1_w       = (const float*)d_in[2];
  const float* ln1_b       = (const float*)d_in[3];
  const float* ln2_w       = (const float*)d_in[4];
  const float* ln2_b       = (const float*)d_in[5];
  const float* w_kern_attn = (const float*)d_in[6];
  const float* rel_table   = (const float*)d_in[7];
  const float* w_kv        = (const float*)d_in[8];
  const float* b_kv        = (const float*)d_in[9];
  const float* w_q         = (const float*)d_in[10];
  const float* b_q         = (const float*)d_in[11];
  const float* w_proj      = (const float*)d_in[12];
  const float* b_proj      = (const float*)d_in[13];
  const float* w_kern_ffn  = (const float*)d_in[14];
  const float* w_pin       = (const float*)d_in[15];
  const float* w_dw        = (const float*)d_in[16];
  const float* w_pout      = (const float*)d_in[17];
  float* out = (float*)d_out;

  char* ws = (char*)d_ws;
  _Float16* xh     = (_Float16*)(ws + OFF_XH);   // also yf for the FFN half
  float*    biasT  = (float*)(ws + OFF_BIAS);
  _Float16* WqT    = (_Float16*)(ws + OFF_WQT);
  float*    bq_pad = (float*)(ws + OFF_BQ);
  _Float16* WkvT   = (_Float16*)(ws + OFF_WKVT);
  float*    bkv_pad= (float*)(ws + OFF_BKV);
  _Float16* WprojT = (_Float16*)(ws + OFF_WPROJT);
  _Float16* WpinT  = (_Float16*)(ws + OFF_WPIN);
  _Float16* WpoutT = (_Float16*)(ws + OFF_WPOUT);
  float*    filmA  = (float*)(ws + OFF_FILMA);
  float*    filmF  = (float*)(ws + OFF_FILMF);

  (void)hipFuncSetAttribute((const void*)attn_kernel,
                            hipFuncAttributeMaxDynamicSharedMemorySize, 294912);
  (void)hipFuncSetAttribute((const void*)ffn_kernel,
                            hipFuncAttributeMaxDynamicSharedMemorySize, 261632);

  prep_kernel<<<256, 256, 0, stream>>>(
      k_v, w_kern_attn, w_kern_ffn, rel_table, w_q, b_q, w_kv, b_kv, w_proj,
      w_pin, w_pout, WqT, bq_pad, WkvT, bkv_pad, WprojT, WpinT, WpoutT, filmA,
      filmF, biasT);

  ln_film_kernel<0><<<1024, 256, 0, stream>>>(x, ln1_w, ln1_b, filmA, xh);

  attn_kernel<<<1024, 256, 294912, stream>>>(xh, WqT, bq_pad, WkvT, bkv_pad,
                                             WprojT, b_proj, biasT, x, out);

  ln_film_kernel<1><<<1024, 256, 0, stream>>>(out, ln2_w, ln2_b, filmF, xh);

  ffn_kernel<<<4096, 256, 261632, stream>>>(xh, WpinT, w_dw, WpoutT, out);
}
